// SimAttn_72447508349113
// MI455X (gfx1250) — matrix operand order repaired and checked by reading, not compiled, not run on hardware
//
#include <hip/hip_runtime.h>
#include <hip/hip_bf16.h>

// ---------------------------------------------------------------------------
// SimAttn on MI455X (gfx1250): bf16 WMMA pipeline
//   proj:  qp/kp/vp = X @ W^T + b   (kp,vp l2-normalized per 64-chunk), bf16
//   kv:    kv[b,h]  = kh^T @ vh     (64x64x4096 per head), stored bf16 [j][i]
//   out:   out      = qh @ kv       (4096x64x64 per head), fp32 merged heads
// ---------------------------------------------------------------------------

typedef __attribute__((ext_vector_type(16))) __bf16 v16bf;
typedef __attribute__((ext_vector_type(8)))  __bf16 bf16x8;
typedef __attribute__((ext_vector_type(4)))  __bf16 bf16x4;
typedef __attribute__((ext_vector_type(8)))  float  v8f;
typedef __attribute__((ext_vector_type(4)))  float  f32x4;

union FragBF16 { v16bf v; bf16x8 h[2]; };

static __device__ __forceinline__ v8f wmma_bf16(v16bf a, v16bf b, v8f c) {
    return __builtin_amdgcn_wmma_f32_16x16x32_bf16(
        /*neg_a=*/false, a, /*neg_b=*/false, b,
        /*c_mod=*/(short)0, c, /*reuse_a=*/false, /*reuse_b=*/false);
}

static __device__ __forceinline__ bf16x4 cvt4(f32x4 v) {
    bf16x4 r;
    r[0] = (__bf16)v[0]; r[1] = (__bf16)v[1];
    r[2] = (__bf16)v[2]; r[3] = (__bf16)v[3];
    return r;
}

// Async global->LDS copy (gfx1250). The builtin takes pointers to int4
// (GCC vector_size(16)) in AS1 (global) / AS3 (LDS).
#if defined(__gfx1250__) && \
    __has_builtin(__builtin_amdgcn_global_load_async_to_lds_b128) && \
    __has_builtin(__builtin_amdgcn_s_wait_asynccnt)
#define USE_ASYNC_LDS 1
typedef int v4i_t __attribute__((vector_size(16)));
typedef __attribute__((address_space(1))) v4i_t* gas_v4i;
typedef __attribute__((address_space(3))) v4i_t* lds_v4i;
static __device__ __forceinline__ gas_v4i as_global(const void* p) {
    return (gas_v4i)(unsigned long long)p;
}
static __device__ __forceinline__ lds_v4i as_lds(void* p) {
    return (lds_v4i)(unsigned int)(unsigned long long)p;
}
#else
#define USE_ASYNC_LDS 0
#endif

#define DMODEL 1024
#define HEADS  16
#define HDIM   64
#define SEQ    4096
#define BATCH  4

// ---------------------------------------------------------------------------
// Kernel 1: Y[m,n] = sum_k X[m,k]*W[n,k] + bias[n]  (bf16 out)
// Block tile 128(M) x 128(N); 8 waves, each 64x32 (4x2 WMMA tiles).
// Double-buffered LDS with register prefetch of the next f32 tile.
// do_norm: fused l2-normalize of each aligned 64-wide output chunk.
// ---------------------------------------------------------------------------
__global__ __launch_bounds__(256)
void proj_kernel(const float* __restrict__ X, const float* __restrict__ W,
                 const float* __restrict__ bias, __bf16* __restrict__ Y,
                 int do_norm)
{
    __shared__ __bf16 As[2][128 * 32];   // [m][k] bf16
    __shared__ __bf16 Bs[2][128 * 32];   // [n][k] bf16 (W rows are [n][k])
    __shared__ float  row_ss[2 * 128];   // [chunk][row] sum of squares

    const int tid  = threadIdx.x;
    const int lane = tid & 31;
    const int wave = tid >> 5;
    const int lo   = lane & 15;
    const int hi   = lane >> 4;
    const int wm   = wave & 1;    // M half (64 rows)
    const int wn   = wave >> 1;   // N quarter (32 cols)
    const long m0  = (long)blockIdx.x * 128;
    const int  n0  = blockIdx.y * 128;

    const v8f vzero = {0.f,0.f,0.f,0.f,0.f,0.f,0.f,0.f};
    v8f acc[4][2];
    #pragma unroll
    for (int mt = 0; mt < 4; ++mt) { acc[mt][0] = vzero; acc[mt][1] = vzero; }

    // --- prologue: stage tile k0=0 (1024 float4 per operand, 4 per thread)
    f32x4 ra[4], rb[4];
    #pragma unroll
    for (int p = 0; p < 4; ++p) {
        int e = tid + p * 256, r = e >> 3, c4 = (e & 7) * 4;
        ra[p] = *(const f32x4*)&X[(m0 + r) * DMODEL + c4];
        rb[p] = *(const f32x4*)&W[(long)(n0 + r) * DMODEL + c4];
    }
    #pragma unroll
    for (int p = 0; p < 4; ++p) {
        int e = tid + p * 256, r = e >> 3, c4 = (e & 7) * 4;
        *(bf16x4*)&As[0][r * 32 + c4] = cvt4(ra[p]);
        *(bf16x4*)&Bs[0][r * 32 + c4] = cvt4(rb[p]);
    }
    __syncthreads();

    int cur = 0;
    for (int k0 = 0; k0 < DMODEL; k0 += 32) {
        const bool more = (k0 + 32) < DMODEL;
        if (more) {
            #pragma unroll
            for (int p = 0; p < 4; ++p) {
                int e = tid + p * 256, r = e >> 3, c4 = (e & 7) * 4;
                ra[p] = *(const f32x4*)&X[(m0 + r) * DMODEL + k0 + 32 + c4];
                rb[p] = *(const f32x4*)&W[(long)(n0 + r) * DMODEL + k0 + 32 + c4];
            }
            if (k0 + 64 < DMODEL)   // L2 prefetch of the tile after next
                __builtin_prefetch(&X[(m0 + (tid >> 1)) * DMODEL + k0 + 64], 0, 0);
        }

        // B fragments (2), then 4 A fragments x 2 WMMA each
        FragBF16 b[2];
        #pragma unroll
        for (int nt = 0; nt < 2; ++nt) {
            const __bf16* bp = &Bs[cur][(wn * 32 + nt * 16 + lo) * 32 + hi * 8];
            b[nt].h[0] = *(const bf16x8*)(bp);
            b[nt].h[1] = *(const bf16x8*)(bp + 16);
        }
        #pragma unroll
        for (int mt = 0; mt < 4; ++mt) {
            FragBF16 a;
            const __bf16* ap = &As[cur][(wm * 64 + mt * 16 + lo) * 32 + hi * 8];
            a.h[0] = *(const bf16x8*)(ap);
            a.h[1] = *(const bf16x8*)(ap + 16);
            acc[mt][0] = wmma_bf16(a.v, b[0].v, acc[mt][0]);
            acc[mt][1] = wmma_bf16(a.v, b[1].v, acc[mt][1]);
        }

        if (more) {
            const int nxt = cur ^ 1;
            #pragma unroll
            for (int p = 0; p < 4; ++p) {
                int e = tid + p * 256, r = e >> 3, c4 = (e & 7) * 4;
                *(bf16x4*)&As[nxt][r * 32 + c4] = cvt4(ra[p]);
                *(bf16x4*)&Bs[nxt][r * 32 + c4] = cvt4(rb[p]);
            }
        }
        __syncthreads();
        cur ^= 1;
    }

    // --- epilogue: bias (+ fused per-64-chunk l2 norm), bf16 store
    float vals[4][2][8];
    #pragma unroll
    for (int mt = 0; mt < 4; ++mt)
        #pragma unroll
        for (int nt = 0; nt < 2; ++nt) {
            float bv = bias[n0 + wn * 32 + nt * 16 + lo];
            #pragma unroll
            for (int j = 0; j < 8; ++j)
                vals[mt][nt][j] = acc[mt][nt][j] + bv;
        }

    if (do_norm) {
        row_ss[tid] = 0.f;                 // 256 entries == 256 threads
        __syncthreads();
        const int chunk = wn >> 1;         // this wave's 64-col chunk
        #pragma unroll
        for (int mt = 0; mt < 4; ++mt)
            #pragma unroll
            for (int j = 0; j < 8; ++j) {
                int r = wm * 64 + mt * 16 + j + 8 * hi;
                float p = vals[mt][0][j] * vals[mt][0][j]
                        + vals[mt][1][j] * vals[mt][1][j];
                atomicAdd(&row_ss[chunk * 128 + r], p);
            }
        __syncthreads();
        #pragma unroll
        for (int mt = 0; mt < 4; ++mt)
            #pragma unroll
            for (int j = 0; j < 8; ++j) {
                int r = wm * 64 + mt * 16 + j + 8 * hi;
                float s = 1.f / fmaxf(sqrtf(row_ss[chunk * 128 + r]), 1e-12f);
                vals[mt][0][j] *= s;
                vals[mt][1][j] *= s;
            }
    }

    #pragma unroll
    for (int mt = 0; mt < 4; ++mt)
        #pragma unroll
        for (int nt = 0; nt < 2; ++nt)
            #pragma unroll
            for (int j = 0; j < 8; ++j) {
                long r = m0 + wm * 64 + mt * 16 + j + 8 * hi;
                int  c = n0 + wn * 32 + nt * 16 + lo;
                Y[r * DMODEL + c] = (__bf16)vals[mt][nt][j];
            }
}

// ---------------------------------------------------------------------------
// Kernel 2: kv[b,h,i,j] = sum_s kh[s,i]*vh[s,j].  One block per (b,h).
// 8 waves: wave owns (i-tile = wave&3, j-tiles (wave>>2)*2+{0,1}).
// Output stored bf16, TRANSPOSED [j][i] so kernel 3 stages it with a
// straight contiguous (async) copy.
// ---------------------------------------------------------------------------
__global__ __launch_bounds__(256)
void kv_kernel(const __bf16* __restrict__ Kp, const __bf16* __restrict__ Vp,
               __bf16* __restrict__ KVT)
{
    __shared__ __bf16 Ks[64 * 32];   // [i][s]
    __shared__ __bf16 Vs[64 * 32];   // [j][s]

    const int tid  = threadIdx.x;
    const int lane = tid & 31;
    const int wave = tid >> 5;
    const int lo   = lane & 15;
    const int hi   = lane >> 4;
    const int it   = wave & 3;
    const int jh   = wave >> 2;
    const int bh   = blockIdx.x;
    const int b    = bh >> 4, h = bh & 15;
    const long base = (long)b * SEQ * DMODEL + h * HDIM;

    const v8f vzero = {0.f,0.f,0.f,0.f,0.f,0.f,0.f,0.f};
    v8f acc[2] = {vzero, vzero};

    for (int s0 = 0; s0 < SEQ; s0 += 32) {
        // stage transposed: [c][s] <- [s][c]
        for (int e = tid; e < 512; e += 256) {
            int c4 = (e & 15) * 4, s = e >> 4;
            bf16x4 tk = *(const bf16x4*)&Kp[base + (long)(s0 + s) * DMODEL + c4];
            bf16x4 tv = *(const bf16x4*)&Vp[base + (long)(s0 + s) * DMODEL + c4];
            #pragma unroll
            for (int u = 0; u < 4; ++u) {
                Ks[(c4 + u) * 32 + s] = tk[u];
                Vs[(c4 + u) * 32 + s] = tv[u];
            }
        }
        __syncthreads();

        FragBF16 a;   // A = kh^T tile: row i, K = s
        const __bf16* ap = &Ks[(it * 16 + lo) * 32 + hi * 8];
        a.h[0] = *(const bf16x8*)(ap);
        a.h[1] = *(const bf16x8*)(ap + 16);
        #pragma unroll
        for (int nt = 0; nt < 2; ++nt) {
            FragBF16 bf;
            const __bf16* bp = &Vs[((jh * 2 + nt) * 16 + lo) * 32 + hi * 8];
            bf.h[0] = *(const bf16x8*)(bp);
            bf.h[1] = *(const bf16x8*)(bp + 16);
            acc[nt] = wmma_bf16(a.v, bf.v, acc[nt]);
        }
        __syncthreads();
    }

    // store bf16 transposed: KVT[bh][j][i]
    #pragma unroll
    for (int nt = 0; nt < 2; ++nt)
        #pragma unroll
        for (int j = 0; j < 8; ++j) {
            int ri = it * 16 + j + 8 * hi;          // i index (C row)
            int cj = (jh * 2 + nt) * 16 + lo;       // j index (C col)
            KVT[(long)bh * HDIM * HDIM + cj * HDIM + ri] = (__bf16)acc[nt][j];
        }
}

// ---------------------------------------------------------------------------
// Kernel 3: out[b,l,h*64+j] = sum_i qp[b,l,h*64+i] * kv[b,h,i,j].
// Grid (L/128, B*H). Wave owns 16 rows x 64 cols (4 WMMA tiles), K=64.
// A fragments load straight from global (b128); kv^T staged into LDS with
// GLOBAL_LOAD_ASYNC_TO_LDS_B128 when available.
// ---------------------------------------------------------------------------
__global__ __launch_bounds__(256)
void out_kernel(const __bf16* __restrict__ Qp, const __bf16* __restrict__ KVT,
                float* __restrict__ Out)
{
    __shared__ __bf16 kvs[64 * 64];   // [j][i] (already transposed in global)

    const int tid  = threadIdx.x;
    const int lane = tid & 31;
    const int wave = tid >> 5;
    const int lo   = lane & 15;
    const int hi   = lane >> 4;
    const int bh   = blockIdx.y;
    const int b    = bh >> 4, h = bh & 15;
    const long l0  = (long)blockIdx.x * 128;
    const long rowbase = (long)b * SEQ + l0;

#if USE_ASYNC_LDS
    for (int e = tid; e < 512; e += 256)
        __builtin_amdgcn_global_load_async_to_lds_b128(
            as_global(&KVT[(long)bh * HDIM * HDIM + e * 8]),
            as_lds(&kvs[e * 8]), 0, 0);
    __builtin_amdgcn_s_wait_asynccnt(0);
    __syncthreads();
#else
    for (int e = tid; e < 512; e += 256)
        *(bf16x8*)&kvs[e * 8] =
            *(const bf16x8*)&KVT[(long)bh * HDIM * HDIM + e * 8];
    __syncthreads();
#endif

    const v8f vzero = {0.f,0.f,0.f,0.f,0.f,0.f,0.f,0.f};
    v8f acc[4] = {vzero, vzero, vzero, vzero};

    const long arow = (rowbase + wave * 16 + lo) * DMODEL + h * HDIM;
    #pragma unroll
    for (int k0 = 0; k0 < HDIM; k0 += 32) {
        FragBF16 a;
        const __bf16* ap = &Qp[arow + k0 + hi * 8];
        a.h[0] = *(const bf16x8*)(ap);
        a.h[1] = *(const bf16x8*)(ap + 16);
        #pragma unroll
        for (int jt = 0; jt < 4; ++jt) {
            FragBF16 bf;
            const __bf16* bp = &kvs[(jt * 16 + lo) * 64 + k0 + hi * 8];
            bf.h[0] = *(const bf16x8*)(bp);
            bf.h[1] = *(const bf16x8*)(bp + 16);
            acc[jt] = wmma_bf16(a.v, bf.v, acc[jt]);
        }
    }

    #pragma unroll
    for (int jt = 0; jt < 4; ++jt)
        #pragma unroll
        for (int j = 0; j < 8; ++j)
            Out[(rowbase + wave * 16 + j + 8 * hi) * DMODEL +
                h * HDIM + jt * 16 + lo] = acc[jt][j];
}

// ---------------------------------------------------------------------------
extern "C" void kernel_launch(void* const* d_in, const int* in_sizes, int n_in,
                              void* d_out, int out_size, void* d_ws, size_t ws_size,
                              hipStream_t stream)
{
    (void)in_sizes; (void)n_in; (void)out_size; (void)ws_size;
    const float* q  = (const float*)d_in[0];
    const float* k  = (const float*)d_in[1];
    const float* v  = (const float*)d_in[2];
    const float* Wq = (const float*)d_in[3];
    const float* bq = (const float*)d_in[4];
    const float* Wk = (const float*)d_in[5];
    const float* bk = (const float*)d_in[6];
    const float* Wv = (const float*)d_in[7];
    const float* bv = (const float*)d_in[8];
    float* out = (float*)d_out;

    const long MT = (long)BATCH * SEQ;           // 16384 rows
    __bf16* qp  = (__bf16*)d_ws;                 // 32 MB
    __bf16* kp  = qp + MT * DMODEL;              // 32 MB
    __bf16* vp  = kp + MT * DMODEL;              // 32 MB
    __bf16* kvt = vp + MT * DMODEL;              // 0.5 MB: [B*H][j][i] bf16

    dim3 blk(256);
    dim3 gproj(MT / 128, DMODEL / 128);
    proj_kernel<<<gproj, blk, 0, stream>>>(q, Wq, bq, qp, 0);
    proj_kernel<<<gproj, blk, 0, stream>>>(k, Wk, bk, kp, 1);
    proj_kernel<<<gproj, blk, 0, stream>>>(v, Wv, bv, vp, 1);
    kv_kernel<<<dim3(BATCH * HEADS), blk, 0, stream>>>(kp, vp, kvt);
    out_kernel<<<dim3(SEQ / 128, BATCH * HEADS), blk, 0, stream>>>(qp, kvt, out);
}
